// BiRWKV_58617713656062
// MI455X (gfx1250) — compile-verified, matrix operand order checked by reading, not run on hardware
//
#include <hip/hip_runtime.h>

typedef _Float16 half_t;
typedef __attribute__((ext_vector_type(16))) _Float16 v16h;
typedef __attribute__((ext_vector_type(8)))  _Float16 v8h;
typedef __attribute__((ext_vector_type(4)))  _Float16 v4h;
typedef __attribute__((ext_vector_type(8)))  float    v8f;

#define T_DIM 2048
#define B_DIM 8
#define I_DIM 1024
#define H_DIM 1024
#define M_DIM (T_DIM * B_DIM)          // 16384 rows of x

union V16U { v16h v; struct { v8h lo, hi; } s; };

// ---------------------------------------------------------------------------
// fp32 -> fp16 conversion (x and the six weight matrices), 4 elems/thread
// ---------------------------------------------------------------------------
__global__ void cvt_f32_to_f16_x4(const float* __restrict__ src,
                                  half_t* __restrict__ dst, int n4) {
  int i = blockIdx.x * blockDim.x + threadIdx.x;
  if (i < n4) {
    const float4 f = ((const float4*)src)[i];
    v4h o = { (half_t)f.x, (half_t)f.y, (half_t)f.z, (half_t)f.w };
    ((v4h*)dst)[i] = o;
  }
}

// ---------------------------------------------------------------------------
// Projection GEMM: [16384 x 1024] x [1024 x 1024]^T for 6 weight matrices.
// Wave computes a 32x32 tile (2x2 WMMA fragments), K in steps of 32.
// A (x rows) and B (weight rows, [N][K] row-major) use the same K-major
// per-lane fragment layout:
//   lanes 0-15 : row = laneL, chunks at K = k0+0..7  and k0+16..23
//   lanes 16-31: row = laneL, chunks at K = k0+8..15 and k0+24..31
// Sigmoid fused into the r-matrix stores (mat % 3 == 0).
// ---------------------------------------------------------------------------
__global__ __launch_bounds__(256)
void proj_gemm_wmma(const half_t* __restrict__ xh,
                    const half_t* __restrict__ wh,
                    float* __restrict__ rkv) {
  const int lane  = threadIdx.x & 31;
  const int wave  = threadIdx.x >> 5;
  const int laneL = lane & 15;
  const int laneH = lane >> 4;

  const int NT_N = (6 * H_DIM) / 32;            // 192 column tiles
  const int tile = blockIdx.x * 8 + wave;
  const int nt   = tile % NT_N;
  const int mt   = tile / NT_N;

  const int m0  = mt * 32;
  const int n0g = nt * 32;
  const int mat = n0g / H_DIM;                  // 0..5: r/k/v fw, r/k/v bw
  const int n0  = n0g % H_DIM;

  const half_t* pa0 = xh + (size_t)(m0 + laneL) * I_DIM;
  const half_t* pa1 = pa0 + (size_t)16 * I_DIM;
  const half_t* pb0 = wh + (size_t)mat * H_DIM * I_DIM
                         + (size_t)(n0 + laneL) * I_DIM;
  const half_t* pb1 = pb0 + (size_t)16 * I_DIM;
  const int koff = laneH * 8;

  v8f c00 = {}, c01 = {}, c10 = {}, c11 = {};

  #pragma unroll 4
  for (int k0 = 0; k0 < I_DIM; k0 += 32) {
    V16U a0, a1, b0, b1;
    a0.s.lo = *(const v8h*)(pa0 + k0 + koff);
    a0.s.hi = *(const v8h*)(pa0 + k0 + 16 + koff);
    a1.s.lo = *(const v8h*)(pa1 + k0 + koff);
    a1.s.hi = *(const v8h*)(pa1 + k0 + 16 + koff);
    b0.s.lo = *(const v8h*)(pb0 + k0 + koff);
    b0.s.hi = *(const v8h*)(pb0 + k0 + 16 + koff);
    b1.s.lo = *(const v8h*)(pb1 + k0 + koff);
    b1.s.hi = *(const v8h*)(pb1 + k0 + 16 + koff);

    c00 = __builtin_amdgcn_wmma_f32_16x16x32_f16(false, a0.v, false, b0.v,
                                                 (short)0, c00, false, false);
    c01 = __builtin_amdgcn_wmma_f32_16x16x32_f16(false, a0.v, false, b1.v,
                                                 (short)0, c01, false, false);
    c10 = __builtin_amdgcn_wmma_f32_16x16x32_f16(false, a1.v, false, b0.v,
                                                 (short)0, c10, false, false);
    c11 = __builtin_amdgcn_wmma_f32_16x16x32_f16(false, a1.v, false, b1.v,
                                                 (short)0, c11, false, false);
  }

  float* dst = rkv + (size_t)mat * M_DIM * H_DIM;
  const bool do_sig = (mat % 3) == 0;           // r matrices get sigmoid

  v8f cs[2][2] = { { c00, c01 }, { c10, c11 } };
  #pragma unroll
  for (int i = 0; i < 2; ++i) {
    #pragma unroll
    for (int j = 0; j < 2; ++j) {
      const int col  = n0 + j * 16 + laneL;
      const int rowb = m0 + i * 16 + laneH * 8;
      #pragma unroll
      for (int v = 0; v < 8; ++v) {
        float val = cs[i][j][v];
        if (do_sig) val = 1.0f / (1.0f + __expf(-val));
        dst[(size_t)(rowb + v) * H_DIM + col] = val;
      }
    }
  }
}

// ---------------------------------------------------------------------------
// Bidirectional RWKV scan: one lane per (dir, b, h) channel; 16384 lanes.
// Writes output [T, B, 2H] (fw in h<1024, bw in h>=1024, bw time-reversed
// back to original order) plus hidden tail [2, B, H].
// ---------------------------------------------------------------------------
__global__ __launch_bounds__(256)
void birwkv_scan(const float* __restrict__ rkv,
                 const float* __restrict__ u_fw, const float* __restrict__ w_fw,
                 const float* __restrict__ u_bw, const float* __restrict__ w_bw,
                 float* __restrict__ out) {
  const int tid = blockIdx.x * blockDim.x + threadIdx.x;  // 0..16383
  const int dir = tid >> 13;
  const int b   = (tid >> 10) & 7;
  const int h   = tid & 1023;

  const size_t MAT = (size_t)M_DIM * H_DIM;
  const float* rp = rkv + (dir ? 3 * MAT : 0);
  const float* kp = rp + MAT;
  const float* vp = kp + MAT;

  const float uu = dir ? u_bw[h] : u_fw[h];
  const float ww = dir ? w_bw[h] : w_fw[h];

  float fn = 0.0f, fd = 0.0f, scale = -1e20f;
  int t        = dir ? (T_DIM - 1) : 0;
  const int dt = dir ? -1 : 1;

  float* hid = out + (size_t)T_DIM * B_DIM * 2 * H_DIM
                   + (size_t)dir * B_DIM * H_DIM + (size_t)b * H_DIM + h;

  for (int s = 0; s < T_DIM; ++s, t += dt) {
    const size_t idx = (size_t)(t * B_DIM + b) * H_DIM + h;
    const float r = rp[idx];
    const float k = kp[idx];
    const float v = vp[idx];

    const float sm = fmaxf(scale, uu + k);
    const float cm = __expf(scale - sm);
    const float am = __expf(uu + k - sm);
    const float y  = r * ((cm * fn + am * v) / (cm * fd + am));

    const float ns  = fmaxf(scale + ww, k);
    const float ncm = __expf(scale + ww - ns);
    const float nam = __expf(k - ns);
    fn = ncm * fn + nam * v;
    fd = ncm * fd + nam;
    scale = ns;

    out[(size_t)(t * B_DIM + b) * (2 * H_DIM) + (size_t)dir * H_DIM + h] = y;
    if (s == T_DIM - 1) *hid = y;
  }
}

// ---------------------------------------------------------------------------
// Host launcher
// ---------------------------------------------------------------------------
extern "C" void kernel_launch(void* const* d_in, const int* in_sizes, int n_in,
                              void* d_out, int out_size, void* d_ws, size_t ws_size,
                              hipStream_t stream) {
  (void)in_sizes; (void)n_in; (void)out_size; (void)ws_size;

  const float* x      = (const float*)d_in[0];
  const float* w_r_fw = (const float*)d_in[1];
  const float* w_k_fw = (const float*)d_in[2];
  const float* w_v_fw = (const float*)d_in[3];
  const float* w_u_fw = (const float*)d_in[4];
  const float* w_w_fw = (const float*)d_in[5];
  const float* w_r_bw = (const float*)d_in[6];
  const float* w_k_bw = (const float*)d_in[7];
  const float* w_v_bw = (const float*)d_in[8];
  const float* w_u_bw = (const float*)d_in[9];
  const float* w_w_bw = (const float*)d_in[10];

  // Workspace layout (bytes):
  //   [0)                xh : f16 x,        T*B*I*2  = 33,554,432
  //   [xh)               wh : f16 weights,  6*H*I*2  = 12,582,912
  //   [xh+wh)            rkv: f32 r/k/v x2, 6*T*B*H*4 = 402,653,184
  const size_t XH_BYTES = (size_t)T_DIM * B_DIM * I_DIM * sizeof(half_t);
  const size_t WH_BYTES = (size_t)6 * H_DIM * I_DIM * sizeof(half_t);
  char*   ws  = (char*)d_ws;
  half_t* xh  = (half_t*)ws;
  half_t* wh  = (half_t*)(ws + XH_BYTES);
  float*  rkv = (float*)(ws + XH_BYTES + WH_BYTES);

  // 1) convert x and weights to f16
  {
    const int n4x = (T_DIM * B_DIM * I_DIM) / 4;
    cvt_f32_to_f16_x4<<<(n4x + 255) / 256, 256, 0, stream>>>(x, xh, n4x);
    const int HI  = H_DIM * I_DIM;
    const int n4w = HI / 4;
    const float* wsrc[6] = { w_r_fw, w_k_fw, w_v_fw, w_r_bw, w_k_bw, w_v_bw };
    for (int m = 0; m < 6; ++m) {
      cvt_f32_to_f16_x4<<<(n4w + 255) / 256, 256, 0, stream>>>(
          wsrc[m], wh + (size_t)m * HI, n4w);
    }
  }

  // 2) WMMA projection GEMMs (r gets fused sigmoid)
  {
    const int tiles  = (M_DIM / 32) * ((6 * H_DIM) / 32);  // 512 * 192 = 98304
    const int blocks = tiles / 8;                          // 8 waves per block
    proj_gemm_wmma<<<blocks, 256, 0, stream>>>(xh, wh, rkv);
  }

  // 3) bidirectional scan -> d_out ([T,B,2H] output + [2,B,H] hidden)
  {
    const int threads = 2 * B_DIM * H_DIM;                 // 16384 channels
    birwkv_scan<<<threads / 256, 256, 0, stream>>>(
        rkv, w_u_fw, w_w_fw, w_u_bw, w_w_bw, (float*)d_out);
  }
}
